// ModifiedGCN_78640851190522
// MI455X (gfx1250) — matrix operand order repaired and checked by reading, not compiled, not run on hardware
//
#include <hip/hip_runtime.h>
#include <hip/hip_bf16.h>

typedef __attribute__((ext_vector_type(16))) _Float16 v16h;
typedef __attribute__((ext_vector_type(8)))  float    v8f;

#define N_GRAPHS  64
#define EDGE_DIM  16
#define OUT_DIM   8
#define BN_EPS    1e-5f

// ---------------------------------------------------------------- zero
__global__ void zero_f32(float* __restrict__ p, long long n) {
    long long i = (long long)blockIdx.x * blockDim.x + threadIdx.x;
    if (i < n) p[i] = 0.0f;
}

// ------------------------------------------------------- degree accumulation
__global__ void degrees_kernel(const float* __restrict__ ew,
                               const int* __restrict__ src,
                               const int* __restrict__ dst,
                               float* __restrict__ out_w, float* __restrict__ in_w,
                               float* __restrict__ od,    float* __restrict__ idg,
                               long long nE) {
    long long e = (long long)blockIdx.x * blockDim.x + threadIdx.x;
    if (e >= nE) return;
    float w = fabsf(ew[e]);
    int s = src[e], d = dst[e];
    atomicAdd(&out_w[s], w);
    atomicAdd(&in_w[d],  w);
    atomicAdd(&od[s], 1.0f);
    atomicAdd(&idg[d], 1.0f);
}

// out_deg/in_deg -> rsqrt(max(deg,1)) in place
__global__ void node_norms(float* __restrict__ od, float* __restrict__ idg, int n) {
    int i = blockIdx.x * blockDim.x + threadIdx.x;
    if (i >= n) return;
    od[i]  = rsqrtf(fmaxf(od[i], 1.0f));
    idg[i] = rsqrtf(fmaxf(idg[i], 1.0f));
}

// wn[e] = |w| * rsqrt(out_w[src] * in_w[dst])
__global__ void wn_kernel(const float* __restrict__ ew,
                          const int* __restrict__ src, const int* __restrict__ dst,
                          const float* __restrict__ out_w, const float* __restrict__ in_w,
                          float* __restrict__ wn, long long nE) {
    long long e = (long long)blockIdx.x * blockDim.x + threadIdx.x;
    if (e >= nE) return;
    float w = fabsf(ew[e]);
    wn[e] = w * rsqrtf(out_w[src[e]] * in_w[dst[e]]);
}

// ------------------------------------------------------- WMMA projection
// h[N, DOUT] = (x[N, DIN] * srcnorm[N, broadcast]) @ W[DIN, DOUT]
// one wave32 per 16x16 output tile; N % 16 == 0 so EXEC is all ones.
template<int DIN, int DOUT>
__global__ __launch_bounds__(32)
void project_wmma(const float* __restrict__ x,
                  const float* __restrict__ srcnorm,
                  const float* __restrict__ W,
                  float* __restrict__ h) {
    const int lane = threadIdx.x & 31;
    const int row0 = blockIdx.x * 16;
    const int n0   = blockIdx.y * 16;
    const int m    = lane & 15;
    const int hi   = lane >> 4;           // 0 or 1
    const int r    = row0 + m;
    const float sn = srcnorm[r];

    v8f c = {};
    for (int k0 = 0; k0 < DIN; k0 += 32) {
        // A: 16-bit A-matrix 16x32 layout (VGPR j<4: K=2j,2j+1 (+8*hi); j>=4: +16)
        v16h a;
#pragma unroll
        for (int i = 0; i < 16; ++i) {
            int k = k0 + ((i < 8) ? 0 : 16) + hi * 8 + (i & 7);
            a[i] = (_Float16)(x[(long long)r * DIN + k] * sn);
        }
        // B: same K order as A (elem i<8: K=8*hi+i; i>=8: K=16+8*hi+(i-8))
        v16h b;
#pragma unroll
        for (int i = 0; i < 16; ++i) {
            int k = k0 + ((i < 8) ? 0 : 16) + hi * 8 + (i & 7);
            b[i] = (_Float16)W[k * DOUT + n0 + m];
        }
        c = __builtin_amdgcn_wmma_f32_16x16x32_f16(
                false, a, false, b, (short)0, c, false, false);
    }
    // D: VGPR rr -> M = rr + 8*hi, N = lane&15
#pragma unroll
    for (int rr = 0; rr < 8; ++rr) {
        int row = row0 + rr + hi * 8;
        h[(long long)row * DOUT + n0 + m] = c[rr];
    }
}

// ------------------------------------------------------- edge message pass
// m[dst] += wn * h[src]; one thread per edge handles all DOUT features
// (reads src/dst/wn exactly once per edge).
template<int DOUT>
__global__ void message_kernel(const float* __restrict__ h,
                               const float* __restrict__ wn,
                               const int* __restrict__ src,
                               const int* __restrict__ dst,
                               float* __restrict__ mout, long long nE) {
    long long e = (long long)blockIdx.x * blockDim.x + threadIdx.x;
    if (e >= nE) return;
    int s = src[e], d = dst[e];
    float w = wn[e];
    const float4* hp = (const float4*)(h + (long long)s * DOUT);
    float* mp = mout + (long long)d * DOUT;
#pragma unroll
    for (int c = 0; c < DOUT / 4; ++c) {
        float4 hv = hp[c];
        atomicAdd(mp + 4 * c + 0, w * hv.x);
        atomicAdd(mp + 4 * c + 1, w * hv.y);
        atomicAdd(mp + 4 * c + 2, w * hv.z);
        atomicAdd(mp + 4 * c + 3, w * hv.w);
    }
}

// ------------------------------------------------------- y = m*dstnorm + b, BN stats
// wave32 tree reduction per feature; one global atomic per wave per feature.
// N % 32 == 0, so guarded-out lanes always form whole waves -> shuffles safe.
template<int DOUT>
__global__ void postconv_stats(const float* __restrict__ m,
                               const float* __restrict__ dstnorm,
                               const float* __restrict__ bias,
                               float* __restrict__ y,
                               float* __restrict__ ssum, float* __restrict__ ssq,
                               int n) {
    int i = blockIdx.x * blockDim.x + threadIdx.x;
    if (i >= n) return;
    const int lane = threadIdx.x & 31;
    float dn = dstnorm[i];
#pragma unroll
    for (int j = 0; j < DOUT; ++j) {
        float v = m[(long long)i * DOUT + j] * dn + bias[j];
        y[(long long)i * DOUT + j] = v;
        float s = v, q = v * v;
#pragma unroll
        for (int off = 16; off > 0; off >>= 1) {
            s += __shfl_xor(s, off);
            q += __shfl_xor(q, off);
        }
        if (lane == 0) {
            atomicAdd(&ssum[j], s);
            atomicAdd(&ssq[j],  q);
        }
    }
}

__global__ void bn_finalize(const float* __restrict__ ssum, const float* __restrict__ ssq,
                            const float* __restrict__ gamma, const float* __restrict__ beta,
                            float* __restrict__ scale, float* __restrict__ shift,
                            int dout, float inv_n) {
    int j = threadIdx.x;
    if (j >= dout) return;
    float mu  = ssum[j] * inv_n;
    float var = ssq[j] * inv_n - mu * mu;
    float sc  = gamma[j] * rsqrtf(var + BN_EPS);
    scale[j] = sc;
    shift[j] = beta[j] - mu * sc;
}

template<int DOUT>
__global__ void bn_apply_elu(const float* __restrict__ y,
                             const float* __restrict__ scale,
                             const float* __restrict__ shift,
                             float* __restrict__ x, long long n) {
    long long i = (long long)blockIdx.x * blockDim.x + threadIdx.x;
    if (i >= n) return;
    int j = (int)(i & (DOUT - 1));      // DOUT is a power of two
    float v = y[i] * scale[j] + shift[j];
    x[i] = v > 0.0f ? v : (__expf(v) - 1.0f);
}

// ------------------------------------------------------- graph pooling
// graph_ids is sorted -> most waves see a single graph; reduce in-wave
// and emit one atomic per feature per wave instead of 32.
__global__ void pool_kernel(const float* __restrict__ x, const int* __restrict__ gid,
                            float* __restrict__ pooled, int n) {
    int i = blockIdx.x * blockDim.x + threadIdx.x;
    if (i >= n) return;                 // N % 32 == 0: whole waves masked
    const int lane = threadIdx.x & 31;
    int g  = gid[i];
    int g0 = __shfl(g, 0);
    if (__all(g == g0)) {
#pragma unroll
        for (int j = 0; j < 16; ++j) {
            float v = x[(long long)i * 16 + j];
#pragma unroll
            for (int off = 16; off > 0; off >>= 1) v += __shfl_xor(v, off);
            if (lane == 0) atomicAdd(&pooled[g0 * 16 + j], v);
        }
    } else {
#pragma unroll
        for (int j = 0; j < 16; ++j)
            atomicAdd(&pooled[g * 16 + j], x[(long long)i * 16 + j]);
    }
}

// edge feature mean: wave tree-reduce -> 1 LDS atomic per wave -> 1 global per block
__global__ void edge_sum_kernel(const float* __restrict__ ef,
                                float* __restrict__ esum, long long nE) {
    __shared__ float ls[EDGE_DIM];
    if (threadIdx.x < EDGE_DIM) ls[threadIdx.x] = 0.f;
    __syncthreads();
    long long e = (long long)blockIdx.x * blockDim.x + threadIdx.x;
    const int lane = threadIdx.x & 31;
#pragma unroll
    for (int j = 0; j < EDGE_DIM; ++j) {
        float v = (e < nE) ? ef[e * EDGE_DIM + j] : 0.0f;
#pragma unroll
        for (int off = 16; off > 0; off >>= 1) v += __shfl_xor(v, off);
        if (lane == 0) atomicAdd(&ls[j], v);
    }
    __syncthreads();
    if (threadIdx.x < EDGE_DIM) atomicAdd(&esum[threadIdx.x], ls[threadIdx.x]);
}

__global__ void final_kernel(const float* __restrict__ pooled,
                             const float* __restrict__ esum,
                             const float* __restrict__ Wc,
                             const float* __restrict__ bc,
                             float* __restrict__ out, float inv_e) {
    int t = threadIdx.x;                 // 512 = 64 graphs x 8 outputs
    int g = t >> 3, o = t & 7;
    float acc = bc[o];
#pragma unroll
    for (int j = 0; j < 16; ++j) acc += pooled[g * 16 + j] * Wc[j * OUT_DIM + o];
#pragma unroll
    for (int k = 0; k < 16; ++k) acc += esum[k] * inv_e * Wc[(16 + k) * OUT_DIM + o];
    out[g * OUT_DIM + o] = acc;
}

// ---------------------------------------------------------------- launch
static inline int gridFor(long long n, int block) {
    return (int)((n + block - 1) / block);
}

extern "C" void kernel_launch(void* const* d_in, const int* in_sizes, int n_in,
                              void* d_out, int out_size, void* d_ws, size_t ws_size,
                              hipStream_t stream) {
    const float* node_f = (const float*)d_in[0];
    const float* edge_f = (const float*)d_in[1];
    const float* ew     = (const float*)d_in[2];
    const int*   src    = (const int*)d_in[3];
    const int*   dst    = (const int*)d_in[4];
    const int*   gid    = (const int*)d_in[5];
    const float* Wl[4]  = {(const float*)d_in[6],  (const float*)d_in[10],
                           (const float*)d_in[14], (const float*)d_in[18]};
    const float* bl[4]  = {(const float*)d_in[7],  (const float*)d_in[11],
                           (const float*)d_in[15], (const float*)d_in[19]};
    const float* gl[4]  = {(const float*)d_in[8],  (const float*)d_in[12],
                           (const float*)d_in[16], (const float*)d_in[20]};
    const float* bel[4] = {(const float*)d_in[9],  (const float*)d_in[13],
                           (const float*)d_in[17], (const float*)d_in[21]};
    const float* Wc = (const float*)d_in[22];
    const float* bc = (const float*)d_in[23];
    float* out = (float*)d_out;

    const int       N = in_sizes[0] / 64;   // 100000
    const long long E = in_sizes[2];        // 3200000

    // workspace layout (floats)
    float* ws     = (float*)d_ws;
    float* out_w  = ws;                          // N
    float* in_w   = out_w + N;                   // N
    float* src_n  = in_w  + N;                   // N (deg -> norm, in place)
    float* dst_n  = src_n + N;                   // N
    float* wn     = dst_n + N;                   // E
    float* hbuf   = wn + E;                      // N*32
    float* mbuf   = hbuf + (size_t)N * 32;       // N*32
    float* ybuf   = mbuf + (size_t)N * 32;       // N*32
    float* xbuf   = ybuf + (size_t)N * 32;       // N*32
    float* ssum   = xbuf + (size_t)N * 32;       // 32
    float* ssq    = ssum + 32;                   // 32
    float* scale  = ssq + 32;                    // 32
    float* shift  = scale + 32;                  // 32
    float* pooled = shift + 32;                  // 64*16
    float* esum   = pooled + (size_t)N_GRAPHS * 16; // 16

    const int BLK = 256;

    // ---- structural + weighted degrees, edge norms
    zero_f32<<<gridFor(4LL * N, BLK), BLK, 0, stream>>>(out_w, 4LL * N);
    degrees_kernel<<<gridFor(E, BLK), BLK, 0, stream>>>(ew, src, dst, out_w, in_w,
                                                        src_n, dst_n, E);
    wn_kernel<<<gridFor(E, BLK), BLK, 0, stream>>>(ew, src, dst, out_w, in_w, wn, E);
    node_norms<<<gridFor(N, BLK), BLK, 0, stream>>>(src_n, dst_n, N);

    const int dout[4] = {32, 32, 32, 16};

    for (int l = 0; l < 4; ++l) {
        const float* xin = (l == 0) ? node_f : xbuf;
        const int Do = dout[l];

        // projection h = (x * src_norm) @ W  (WMMA f16->f32)
        dim3 pg(N / 16, Do / 16);
        if (l == 0)      project_wmma<64, 32><<<pg, 32, 0, stream>>>(xin, src_n, Wl[l], hbuf);
        else if (l < 3)  project_wmma<32, 32><<<pg, 32, 0, stream>>>(xin, src_n, Wl[l], hbuf);
        else             project_wmma<32, 16><<<pg, 32, 0, stream>>>(xin, src_n, Wl[l], hbuf);

        // edge aggregation into mbuf (one thread per edge)
        zero_f32<<<gridFor((long long)N * Do, BLK), BLK, 0, stream>>>(mbuf, (long long)N * Do);
        if (Do == 32) message_kernel<32><<<gridFor(E, BLK), BLK, 0, stream>>>(
                          hbuf, wn, src, dst, mbuf, E);
        else          message_kernel<16><<<gridFor(E, BLK), BLK, 0, stream>>>(
                          hbuf, wn, src, dst, mbuf, E);

        // y = m*dst_norm + b ; BN stats via wave reductions
        zero_f32<<<1, 64, 0, stream>>>(ssum, 64);   // ssum+ssq contiguous
        if (Do == 32) postconv_stats<32><<<gridFor(N, BLK), BLK, 0, stream>>>(
                          mbuf, dst_n, bl[l], ybuf, ssum, ssq, N);
        else          postconv_stats<16><<<gridFor(N, BLK), BLK, 0, stream>>>(
                          mbuf, dst_n, bl[l], ybuf, ssum, ssq, N);
        bn_finalize<<<1, 32, 0, stream>>>(ssum, ssq, gl[l], bel[l], scale, shift,
                                          Do, 1.0f / (float)N);
        if (Do == 32) bn_apply_elu<32><<<gridFor((long long)N * Do, BLK), BLK, 0, stream>>>(
                          ybuf, scale, shift, xbuf, (long long)N * Do);
        else          bn_apply_elu<16><<<gridFor((long long)N * Do, BLK), BLK, 0, stream>>>(
                          ybuf, scale, shift, xbuf, (long long)N * Do);
    }

    // ---- pooling + edge mean + classifier head
    zero_f32<<<1, 1040, 0, stream>>>(pooled, (long long)N_GRAPHS * 16 + EDGE_DIM);
    pool_kernel<<<gridFor(N, BLK), BLK, 0, stream>>>(xbuf, gid, pooled, N);
    edge_sum_kernel<<<gridFor(E, BLK), BLK, 0, stream>>>(edge_f, esum, E);
    final_kernel<<<1, 512, 0, stream>>>(pooled, esum, Wc, bc, out, 1.0f / (float)E);
}
